// SimpleRoutedExperts_90245852823884
// MI455X (gfx1250) — compile-verified, matrix operand order checked by reading, not run on hardware
//
#include <hip/hip_runtime.h>
#include <hip/hip_bf16.h>

// ---------------- problem constants ----------------
#define T_NUM 16384
#define D_DIM 2048
#define H_DIM 1408
#define E_NUM 16
#define TOPK  4

#define BM 128
#define BK 32
#define MAX_TILES 528                    // 65536/128 + 16 (per-expert pad)
#define MAX_ROWS  (MAX_TILES * BM)       // 67584 padded gathered rows

typedef __bf16 bf16_t;
typedef __attribute__((ext_vector_type(16))) __bf16 v16bf;
typedef __attribute__((ext_vector_type(8)))  __bf16 v8bf;
typedef __attribute__((ext_vector_type(4)))  __bf16 v4bf;
typedef __attribute__((ext_vector_type(8)))  float  v8f;

__device__ __forceinline__ bf16_t f2bf(float f) { return (bf16_t)f; }

// ---------------- init / routing kernels ----------------
__global__ void zero_out_kernel(float* __restrict__ y, int n4) {
  int i = blockIdx.x * blockDim.x + threadIdx.x;
  if (i < n4) ((float4*)y)[i] = make_float4(0.f, 0.f, 0.f, 0.f);
}

__global__ void route_init_kernel(int* counts, int* ids, float* cw) {
  int i = blockIdx.x * blockDim.x + threadIdx.x;
  if (i < MAX_ROWS) { ids[i] = -1; cw[i] = 0.f; }
  if (i < E_NUM) counts[i] = 0;
}

__global__ void route_hist_kernel(const int* __restrict__ indices, int* counts) {
  int i = blockIdx.x * blockDim.x + threadIdx.x;
  if (i < T_NUM * TOPK) atomicAdd(&counts[indices[i] & (E_NUM - 1)], 1);
}

__global__ void route_setup_kernel(const int* __restrict__ counts,
                                   int* offs, int* cursors, int* tilemap) {
  if (threadIdx.x != 0) return;
  int o = 0;
  for (int e = 0; e < E_NUM; ++e) {
    offs[e] = o;
    cursors[e] = o;
    o += (counts[e] + BM - 1) & ~(BM - 1);
  }
  offs[E_NUM] = o;
  int tiles = o >> 7;
  for (int t = 0; t < MAX_TILES; ++t) {
    int e = -1;
    if (t < tiles) {
      int r = t << 7;
      for (int j = 0; j < E_NUM; ++j)
        if (r >= offs[j] && r < offs[j + 1]) { e = j; break; }
    }
    tilemap[t] = e;
  }
}

__global__ void route_scatter_kernel(const int* __restrict__ indices,
                                     const float* __restrict__ weights,
                                     int* cursors, int* ids, float* cw) {
  int i = blockIdx.x * blockDim.x + threadIdx.x;
  if (i >= T_NUM * TOPK) return;
  int e = indices[i] & (E_NUM - 1);
  int pos = atomicAdd(&cursors[e], 1);
  ids[pos] = i >> 2;           // token id (i = t*TOPK + k)
  cw[pos]  = weights[i];
}

// ---------------- GEMM1: h = x @ W1[e]^T, act = u * silu(g) ----------------
// grid.x = MAX_TILES (128-row padded segments), grid.y = H/64 (64 act cols).
// Block: 128x128 of h. LDS B-row n -> W1 row: g0=n>>6, t=n&63,
//   t<32 -> u row (jb + g0*32 + t), t>=32 -> g row (H + jb + g0*32 + t-32)
// so each wave's 64-col group holds matching u (tiles 0-1) and g (tiles 2-3).
// Wave tile: 2 M-tiles x 4 N-tiles. Software-pipelined, double-buffered LDS.
__global__ __launch_bounds__(256)
void moe_gemm1_kernel(const float* __restrict__ x,
                      const float* __restrict__ W1,
                      const int* __restrict__ tilemap,
                      const int* __restrict__ ids,
                      bf16_t* __restrict__ act) {
  const int e = tilemap[blockIdx.x];
  if (e < 0) return;                       // block-uniform: EXEC all-1 at WMMA
  const int jb = blockIdx.y * 64;
  const int rowBase = blockIdx.x * BM;

  __shared__ __align__(16) bf16_t As[2][BM][40];  // 80B row stride (16B aligned)
  __shared__ __align__(16) bf16_t Bs[2][BM][40];

  const int tid  = threadIdx.x;
  const int lane = tid & 31;
  const int wid  = tid >> 5;
  const int sel  = lane >> 4;              // half-wave: K phase select
  const int lrow = lane & 15;
  const int wrow = wid & 3;                // wave M group (32 rows)
  const int wcol = wid >> 2;               // wave N group (64 cols)
  const int mA = wrow * 32;
  const int nB = wcol * 64;

  const float* W1e = W1 + (size_t)e * (2 * H_DIM) * D_DIM;

  // loader mapping: 128 rows x 32 fp32, float4 chunks; 32 rows per pass
  const int r0 = tid >> 3, c4 = tid & 7;
  const float* aptr[4];
  float amask[4];
  const float* bptr[4];
  #pragma unroll
  for (int rr = 0; rr < 4; ++rr) {
    int row = r0 + 32 * rr;
    int tok = ids[rowBase + row];
    amask[rr] = (tok >= 0) ? 1.f : 0.f;    // branchless invalid-row handling
    int tok0  = (tok >= 0) ? tok : 0;
    aptr[rr] = x + (size_t)tok0 * D_DIM + c4 * 4;
    int g0 = row >> 6, t = row & 63;       // interleaved u/g layout (see above)
    int w1row = (t < 32) ? (jb + g0 * 32 + t)
                         : (H_DIM + jb + g0 * 32 + (t - 32));
    bptr[rr] = W1e + (size_t)w1row * D_DIM + c4 * 4;
  }

  v8f acc[2][4] = {};
  float4 aReg[4], bReg[4];

  // stage 0 prefetch
  #pragma unroll
  for (int rr = 0; rr < 4; ++rr) {
    aReg[rr] = *(const float4*)(aptr[rr]);
    bReg[rr] = *(const float4*)(bptr[rr]);
  }

  int buf = 0;
  for (int k0 = 0; k0 < D_DIM; k0 += BK, buf ^= 1) {
    // convert + store current stage
    #pragma unroll
    for (int rr = 0; rr < 4; ++rr) {
      int row = r0 + 32 * rr;
      v4bf a4;
      a4[0] = f2bf(aReg[rr].x * amask[rr]);
      a4[1] = f2bf(aReg[rr].y * amask[rr]);
      a4[2] = f2bf(aReg[rr].z * amask[rr]);
      a4[3] = f2bf(aReg[rr].w * amask[rr]);
      *(v4bf*)&As[buf][row][c4 * 4] = a4;
      v4bf b4;
      b4[0] = f2bf(bReg[rr].x);
      b4[1] = f2bf(bReg[rr].y);
      b4[2] = f2bf(bReg[rr].z);
      b4[3] = f2bf(bReg[rr].w);
      *(v4bf*)&Bs[buf][row][c4 * 4] = b4;
    }
    // prefetch next stage while this one computes
    if (k0 + BK < D_DIM) {
      #pragma unroll
      for (int rr = 0; rr < 4; ++rr) {
        aReg[rr] = *(const float4*)(aptr[rr] + k0 + BK);
        bReg[rr] = *(const float4*)(bptr[rr] + k0 + BK);
      }
    }
    __syncthreads();

    // fragments: lane<16 holds K 0..7 & 16..23, lane>=16 holds K 8..15 & 24..31
    v16bf afr[2];
    #pragma unroll
    for (int mi = 0; mi < 2; ++mi) {
      v8bf lo = *(const v8bf*)&As[buf][mA + mi * 16 + lrow][sel * 8];
      v8bf hi = *(const v8bf*)&As[buf][mA + mi * 16 + lrow][16 + sel * 8];
      #pragma unroll
      for (int i = 0; i < 8; ++i) { afr[mi][i] = lo[i]; afr[mi][i + 8] = hi[i]; }
    }
    v16bf bfr[4];
    #pragma unroll
    for (int ni = 0; ni < 4; ++ni) {
      v8bf lo = *(const v8bf*)&Bs[buf][nB + ni * 16 + lrow][sel * 8];
      v8bf hi = *(const v8bf*)&Bs[buf][nB + ni * 16 + lrow][16 + sel * 8];
      #pragma unroll
      for (int i = 0; i < 8; ++i) { bfr[ni][i] = lo[i]; bfr[ni][i + 8] = hi[i]; }
    }
    #pragma unroll
    for (int mi = 0; mi < 2; ++mi)
      #pragma unroll
      for (int ni = 0; ni < 4; ++ni)
        acc[mi][ni] = __builtin_amdgcn_wmma_f32_16x16x32_bf16(
            false, afr[mi], false, bfr[ni], (short)0, acc[mi][ni], false, false);
  }

  // epilogue: act = u * g * sigmoid(g); tiles ni(0..1)=u pair with ni+2=g
  #pragma unroll
  for (int mi = 0; mi < 2; ++mi) {
    #pragma unroll
    for (int ni = 0; ni < 2; ++ni) {
      #pragma unroll
      for (int i = 0; i < 8; ++i) {
        float u = acc[mi][ni][i];
        float g = acc[mi][ni + 2][i];
        float s = __builtin_amdgcn_rcpf(1.0f + __builtin_amdgcn_exp2f(-1.44269504089f * g));
        float v = u * g * s;
        int row = rowBase + mA + mi * 16 + i + (sel << 3);
        int col = jb + wcol * 32 + ni * 16 + lrow;
        act[(size_t)row * H_DIM + col] = f2bf(v);
      }
    }
  }
}

// ---------------- GEMM2: y[tok] += cw * (act @ W2[e]^T) ----------------
// grid.x = MAX_TILES, grid.y = D/128. Same pipelined 2x4 wave-tile structure.
__global__ __launch_bounds__(256)
void moe_gemm2_kernel(const bf16_t* __restrict__ act,
                      const float* __restrict__ W2,
                      const int* __restrict__ tilemap,
                      const int* __restrict__ ids,
                      const float* __restrict__ cw,
                      float* __restrict__ y) {
  const int e = tilemap[blockIdx.x];
  if (e < 0) return;
  const int nb = blockIdx.y * 128;
  const int rowBase = blockIdx.x * BM;

  __shared__ __align__(16) bf16_t As[2][BM][40];
  __shared__ __align__(16) bf16_t Bs[2][BM][40];

  const int tid  = threadIdx.x;
  const int lane = tid & 31;
  const int wid  = tid >> 5;
  const int sel  = lane >> 4;
  const int lrow = lane & 15;
  const int wrow = wid & 3;
  const int wcol = wid >> 2;
  const int mA = wrow * 32;
  const int nB = wcol * 64;

  const float* W2e = W2 + (size_t)e * D_DIM * H_DIM;

  const int r0a = tid >> 2, ca = tid & 3;   // act: 64 rows/pass, 16B bf16 chunks
  const int r0b = tid >> 3, cb = tid & 7;   // W2 : 32 rows/pass, float4 chunks

  const bf16_t* asrc[2];
  #pragma unroll
  for (int rr = 0; rr < 2; ++rr)
    asrc[rr] = act + (size_t)(rowBase + r0a + 64 * rr) * H_DIM + ca * 8;
  const float* bsrc[4];
  #pragma unroll
  for (int rr = 0; rr < 4; ++rr)
    bsrc[rr] = W2e + (size_t)(nb + r0b + 32 * rr) * H_DIM + cb * 4;

  v8f acc[2][4] = {};
  uint4  aReg[2];
  float4 bReg[4];

  #pragma unroll
  for (int rr = 0; rr < 2; ++rr) aReg[rr] = *(const uint4*)(asrc[rr]);
  #pragma unroll
  for (int rr = 0; rr < 4; ++rr) bReg[rr] = *(const float4*)(bsrc[rr]);

  int buf = 0;
  for (int k0 = 0; k0 < H_DIM; k0 += BK, buf ^= 1) {
    #pragma unroll
    for (int rr = 0; rr < 2; ++rr)
      *(uint4*)&As[buf][r0a + 64 * rr][ca * 8] = aReg[rr];
    #pragma unroll
    for (int rr = 0; rr < 4; ++rr) {
      v4bf b4;
      b4[0] = f2bf(bReg[rr].x);
      b4[1] = f2bf(bReg[rr].y);
      b4[2] = f2bf(bReg[rr].z);
      b4[3] = f2bf(bReg[rr].w);
      *(v4bf*)&Bs[buf][r0b + 32 * rr][cb * 4] = b4;
    }
    if (k0 + BK < H_DIM) {
      #pragma unroll
      for (int rr = 0; rr < 2; ++rr) aReg[rr] = *(const uint4*)(asrc[rr] + k0 + BK);
      #pragma unroll
      for (int rr = 0; rr < 4; ++rr) bReg[rr] = *(const float4*)(bsrc[rr] + k0 + BK);
    }
    __syncthreads();

    v16bf afr[2];
    #pragma unroll
    for (int mi = 0; mi < 2; ++mi) {
      v8bf lo = *(const v8bf*)&As[buf][mA + mi * 16 + lrow][sel * 8];
      v8bf hi = *(const v8bf*)&As[buf][mA + mi * 16 + lrow][16 + sel * 8];
      #pragma unroll
      for (int i = 0; i < 8; ++i) { afr[mi][i] = lo[i]; afr[mi][i + 8] = hi[i]; }
    }
    v16bf bfr[4];
    #pragma unroll
    for (int ni = 0; ni < 4; ++ni) {
      v8bf lo = *(const v8bf*)&Bs[buf][nB + ni * 16 + lrow][sel * 8];
      v8bf hi = *(const v8bf*)&Bs[buf][nB + ni * 16 + lrow][16 + sel * 8];
      #pragma unroll
      for (int i = 0; i < 8; ++i) { bfr[ni][i] = lo[i]; bfr[ni][i + 8] = hi[i]; }
    }
    #pragma unroll
    for (int mi = 0; mi < 2; ++mi)
      #pragma unroll
      for (int ni = 0; ni < 4; ++ni)
        acc[mi][ni] = __builtin_amdgcn_wmma_f32_16x16x32_bf16(
            false, afr[mi], false, bfr[ni], (short)0, acc[mi][ni], false, false);
  }

  // epilogue: scale by routing weight, scatter-add to y
  #pragma unroll
  for (int mi = 0; mi < 2; ++mi) {
    const int rbase = rowBase + mA + mi * 16 + (sel << 3);
    int   tok8[8];
    float cw8[8];
    #pragma unroll
    for (int i = 0; i < 8; ++i) { tok8[i] = ids[rbase + i]; cw8[i] = cw[rbase + i]; }
    #pragma unroll
    for (int ni = 0; ni < 4; ++ni) {
      #pragma unroll
      for (int i = 0; i < 8; ++i) {
        if (tok8[i] < 0) continue;
        float v = acc[mi][ni][i] * cw8[i];
        atomicAdd(&y[(size_t)tok8[i] * D_DIM + nb + nB + ni * 16 + lrow], v);
      }
    }
  }
}

// ---------------- host launch ----------------
extern "C" void kernel_launch(void* const* d_in, const int* in_sizes, int n_in,
                              void* d_out, int out_size, void* d_ws, size_t ws_size,
                              hipStream_t stream) {
  (void)in_sizes; (void)n_in; (void)out_size; (void)ws_size;
  const float* x       = (const float*)d_in[0];
  const float* weights = (const float*)d_in[1];
  const int*   indices = (const int*)d_in[2];
  const float* W1      = (const float*)d_in[3];
  const float* W2      = (const float*)d_in[4];
  float* y = (float*)d_out;

  char* w = (char*)d_ws;
  int*    counts  = (int*)w;                       // [16]
  int*    cursors = counts + 16;                   // [16]
  int*    offs    = counts + 32;                   // [17]
  int*    tilemap = counts + 64;                   // [528]
  int*    ids     = counts + 1024;                 // byte 4096, [MAX_ROWS]
  float*  cwbuf   = (float*)(w + 4096 + (size_t)MAX_ROWS * 4);
  bf16_t* act     = (bf16_t*)(w + 4096 + (size_t)MAX_ROWS * 8);  // [MAX_ROWS*H] bf16

  zero_out_kernel<<<(T_NUM * D_DIM / 4 + 255) / 256, 256, 0, stream>>>(y, T_NUM * D_DIM / 4);
  route_init_kernel<<<(MAX_ROWS + 255) / 256, 256, 0, stream>>>(counts, ids, cwbuf);
  route_hist_kernel<<<(T_NUM * TOPK) / 256, 256, 0, stream>>>(indices, counts);
  route_setup_kernel<<<1, 32, 0, stream>>>(counts, offs, cursors, tilemap);
  route_scatter_kernel<<<(T_NUM * TOPK) / 256, 256, 0, stream>>>(indices, weights, cursors, ids, cwbuf);
  moe_gemm1_kernel<<<dim3(MAX_TILES, H_DIM / 64), 256, 0, stream>>>(x, W1, tilemap, ids, act);
  moe_gemm2_kernel<<<dim3(MAX_TILES, D_DIM / 128), 256, 0, stream>>>(act, W2, tilemap, ids, cwbuf, y);
}
